// DWTAttention_25744033972847
// MI455X (gfx1250) — matrix operand order repaired and checked by reading, not compiled, not run on hardware
//
#include <hip/hip_runtime.h>
#include <hip/hip_bf16.h>
#include <cstdint>

typedef __attribute__((ext_vector_type(16))) __bf16        v16bf;
typedef __attribute__((ext_vector_type(8)))  __bf16        v8bf;
typedef __attribute__((ext_vector_type(8)))  float         v8f;
typedef __attribute__((ext_vector_type(4)))  int           v4i;
typedef __attribute__((ext_vector_type(8)))  unsigned short v8us;
typedef __attribute__((address_space(3)))    v4i           lds_v4i;

// ---------------- CDNA5 async copy to LDS ----------------
#if defined(__has_builtin)
#if __has_builtin(__builtin_amdgcn_global_load_async_to_lds_b128)
#define USE_ASYNC_COPY 1
#endif
#endif
#ifndef USE_ASYNC_COPY
#define USE_ASYNC_COPY 0
#endif

__device__ __forceinline__ void cp16(const void* g, void* l) {
#if USE_ASYNC_COPY
  v4i* gp = (v4i*)const_cast<void*>(g);
  lds_v4i* lp = (lds_v4i*)(unsigned int)(uintptr_t)l;   // low 32 bits == LDS byte offset
  __builtin_amdgcn_global_load_async_to_lds_b128(gp, lp, 0, 0);
#else
  *reinterpret_cast<v4i*>(l) = *reinterpret_cast<const v4i*>(g);
#endif
}

__device__ __forceinline__ void async_wait() {
#if USE_ASYNC_COPY
#if defined(__has_builtin) && __has_builtin(__builtin_amdgcn_s_wait_asynccnt)
  __builtin_amdgcn_s_wait_asynccnt(0);
#else
  asm volatile("s_wait_asynccnt 0" ::: "memory");
#endif
#endif
}

// f32 <-> bf16 via pure bit ops (RNE), no libcall/ISA-selection risk
__device__ __forceinline__ unsigned short f2bf_u16(float f) {
  unsigned int u = __builtin_bit_cast(unsigned int, f);
  return (unsigned short)((u + 0x7FFFu + ((u >> 16) & 1u)) >> 16);
}
__device__ __forceinline__ __bf16 f2bf(float f) {
  return __builtin_bit_cast(__bf16, f2bf_u16(f));
}
__device__ __forceinline__ float bf2f(__bf16 b) {
  unsigned int u = ((unsigned int)__builtin_bit_cast(unsigned short, b)) << 16;
  return __builtin_bit_cast(float, u);
}

__device__ __forceinline__ void store_out(float* p, float v)  { *p = v; }
__device__ __forceinline__ void store_out(__bf16* p, float v) { *p = f2bf(v); }

// ---------------- one-shot f32 -> bf16 conversion (memory-bound, 8 elems/thread) ----------------
__global__ __launch_bounds__(256) void cvt_f32_to_bf16_x8(
    const float* __restrict__ src, __bf16* __restrict__ dst)
{
  const size_t i = ((size_t)blockIdx.x * 256 + threadIdx.x) * 8;
  const float4 a = *reinterpret_cast<const float4*>(src + i);
  const float4 b = *reinterpret_cast<const float4*>(src + i + 4);
  v8us p;
  p[0] = f2bf_u16(a.x); p[1] = f2bf_u16(a.y); p[2] = f2bf_u16(a.z); p[3] = f2bf_u16(a.w);
  p[4] = f2bf_u16(b.x); p[5] = f2bf_u16(b.y); p[6] = f2bf_u16(b.z); p[7] = f2bf_u16(b.w);
  *reinterpret_cast<v8us*>(dst + i) = p;
}

// ---------------- GEMM: C[N x ncols] = A[N x K] * W[ncols x K]^T + bias ----------------
// A, W are bf16 in global; 256 threads = 8 waves; C tile 128x128; wave tile 32(M) x 64(N).
// Hot loop: async b128 tile fill -> ds_load_b128 fragments -> 8x v_wmma per K-step. No cvt VALU.
template <typename TOUT>
__global__ __launch_bounds__(256) void dwt_gemm_bf16(
    const __bf16* __restrict__ A, const __bf16* __restrict__ W,
    const float* __restrict__ bias, TOUT* __restrict__ C,
    int K, int ncols)
{
  constexpr int LSTR = 40; // halfs/row: 80B, 16B-aligned, bank-spread
  __shared__ __align__(16) __bf16 As[128 * LSTR];
  __shared__ __align__(16) __bf16 Bs[128 * LSTR];

  const int tid  = threadIdx.x;
  const int lane = tid & 31;
  const int wid  = tid >> 5;
  const int wm   = wid & 3;   // 4 waves along M
  const int wn   = wid >> 2;  // 2 waves along N
  const int row0 = blockIdx.y * 128;
  const int col0 = blockIdx.x * 128;
  const int hl   = lane >> 4;
  const int l16  = lane & 15;

  union FragU { v16bf w; v8bf h[2]; };

  v8f acc[2][4] = {};

  for (int k0 = 0; k0 < K; k0 += 32) {
    // Stage 128x32 bf16 tiles of A and W into LDS (512 x 16B chunks each)
#pragma unroll
    for (int j = 0; j < 2; ++j) {
      const int linear = tid + j * 256;
      const int r = linear >> 2;        // 4 chunks per 32-half row
      const int c = (linear & 3) * 8;   // half offset in row
      cp16(A + (size_t)(row0 + r) * K + k0 + c, &As[r * LSTR + c]);
      cp16(W + (size_t)(col0 + r) * K + k0 + c, &Bs[r * LSTR + c]);
    }
    async_wait();
    __syncthreads();

    if (k0 + 32 < K) { // prefetch next K-step tiles
      const int r = tid & 127;
      const int c = (tid >> 7) * 16;
      __builtin_prefetch(A + (size_t)(row0 + r) * K + k0 + 32 + c, 0, 1);
      __builtin_prefetch(W + (size_t)(col0 + r) * K + k0 + 32 + c, 0, 1);
    }

    // A fragments: 16-bit A 16x32 — lane<16: K {0..7,16..23}; lane>=16: K {8..15,24..31}
    v16bf afrag[2];
#pragma unroll
    for (int mi = 0; mi < 2; ++mi) {
      const __bf16* p = &As[(wm * 32 + mi * 16 + l16) * LSTR + hl * 8];
      FragU u;
      u.h[0] = *reinterpret_cast<const v8bf*>(p);
      u.h[1] = *reinterpret_cast<const v8bf*>(p + 16);
      afrag[mi] = u.w;
    }
    // B fragments: 16-bit B 32x16 — lane<16: K {0..7,16..23}; lane>=16: K {8..15,24..31}
    v16bf bfrag[4];
#pragma unroll
    for (int ni = 0; ni < 4; ++ni) {
      const __bf16* p = &Bs[(wn * 64 + ni * 16 + l16) * LSTR + hl * 8];
      FragU u;
      u.h[0] = *reinterpret_cast<const v8bf*>(p);
      u.h[1] = *reinterpret_cast<const v8bf*>(p + 16);
      bfrag[ni] = u.w;
    }

#pragma unroll
    for (int mi = 0; mi < 2; ++mi)
#pragma unroll
      for (int ni = 0; ni < 4; ++ni)
        acc[mi][ni] = __builtin_amdgcn_wmma_f32_16x16x32_bf16(
            false, afrag[mi], false, bfrag[ni], (short)0, acc[mi][ni], false, false);

    __syncthreads();
  }

  // Epilogue: C/D 16x16 f32 layout: VGPR g -> M = g + 8*(lane>=16), N = lane&15
#pragma unroll
  for (int mi = 0; mi < 2; ++mi)
#pragma unroll
    for (int ni = 0; ni < 4; ++ni) {
      const int n    = col0 + wn * 64 + ni * 16 + l16;
      const float bb = bias[n];
      const int m0   = row0 + wm * 32 + mi * 16 + hl * 8;
      TOUT* cp = C + (size_t)m0 * ncols + n;
#pragma unroll
      for (int g = 0; g < 8; ++g)
        store_out(cp + (size_t)g * ncols, acc[mi][ni][g] + bb);
    }
}

// ---------------- 3-level Haar DWT -> weighted product -> inverse, per 8-sample group ----------------
// q/v are bf16 [B, L, D], D = H*Dh (d = h*64 + dh). f32 math, in-place into qb.
__global__ __launch_bounds__(256) void dwt_haar_mix(
    __bf16* __restrict__ qb, const __bf16* __restrict__ vb,
    const float* __restrict__ wq, const float* __restrict__ wv)
{
  constexpr float S  = 0.70710678118654752440f;
  constexpr int  Dc  = 1024, Lc = 4096;
  const size_t gid = (size_t)blockIdx.x * 256 + threadIdx.x; // 2^22 total
  const int dh = (int)(gid & 63);
  const int g  = (int)((gid >> 6) & 511);
  const int h  = (int)((gid >> 15) & 15);
  const int b  = (int)(gid >> 19);

  const size_t base = ((size_t)b * Lc + (size_t)g * 8) * Dc + (size_t)h * 64 + dh;

  float x[8], y[8];
#pragma unroll
  for (int r = 0; r < 8; ++r) {
    x[r] = bf2f(qb[base + (size_t)r * Dc]);
    y[r] = bf2f(vb[base + (size_t)r * Dc]);
  }

  const float wq0 = wq[(h * 3 + 0) * 64 + dh];
  const float wq1 = wq[(h * 3 + 1) * 64 + dh];
  const float wq2 = wq[(h * 3 + 2) * 64 + dh];
  const float wv0 = wv[(h * 3 + 0) * 64 + dh];
  const float wv1 = wv[(h * 3 + 1) * 64 + dh];
  const float wv2 = wv[(h * 3 + 2) * 64 + dh];

  float a1q[4], d1q[4], a1v[4], d1v[4];
#pragma unroll
  for (int i = 0; i < 4; ++i) {
    a1q[i] = (x[2 * i] + x[2 * i + 1]) * S;  d1q[i] = (x[2 * i] - x[2 * i + 1]) * S;
    a1v[i] = (y[2 * i] + y[2 * i + 1]) * S;  d1v[i] = (y[2 * i] - y[2 * i + 1]) * S;
  }
  const float a2q0 = (a1q[0] + a1q[1]) * S, a2q1 = (a1q[2] + a1q[3]) * S;
  const float d2q0 = (a1q[0] - a1q[1]) * S, d2q1 = (a1q[2] - a1q[3]) * S;
  const float a2v0 = (a1v[0] + a1v[1]) * S, a2v1 = (a1v[2] + a1v[3]) * S;
  const float d2v0 = (a1v[0] - a1v[1]) * S, d2v1 = (a1v[2] - a1v[3]) * S;
  const float a3q = (a2q0 + a2q1) * S, d3q = (a2q0 - a2q1) * S;
  const float a3v = (a2v0 + a2v1) * S, d3v = (a2v0 - a2v1) * S;

  const float p0 = (a3q * wq0) * (a3v * wv0);            // cA3 term
  const float p1 = (d3q * wq1) * (d3v * wv1);            // cD3 term
  const float p2[2] = { (d2q0 * wq2) * (d2v0 * wv2),
                        (d2q1 * wq2) * (d2v1 * wv2) };   // cD2 terms
  float p3[4];
#pragma unroll
  for (int i = 0; i < 4; ++i) p3[i] = d1q[i] * d1v[i];   // cD1 terms (no weight)

  const float A2[2] = { (p0 + p1) * S, (p0 - p1) * S };
  const float A1[4] = { (A2[0] + p2[0]) * S, (A2[0] - p2[0]) * S,
                        (A2[1] + p2[1]) * S, (A2[1] - p2[1]) * S };
#pragma unroll
  for (int i = 0; i < 4; ++i) {
    qb[base + (size_t)(2 * i)     * Dc] = f2bf((A1[i] + p3[i]) * S);
    qb[base + (size_t)(2 * i + 1) * Dc] = f2bf((A1[i] - p3[i]) * S);
  }
}

// ---------------- launch ----------------
extern "C" void kernel_launch(void* const* d_in, const int* in_sizes, int n_in,
                              void* d_out, int out_size, void* d_ws, size_t ws_size,
                              hipStream_t stream) {
  (void)in_sizes; (void)n_in; (void)out_size; (void)ws_size;
  const float* query = (const float*)d_in[0];  // [8,4096,1024]
  const float* W_qkv = (const float*)d_in[1];  // [3072,1024]
  const float* b_qkv = (const float*)d_in[2];  // [3072]
  const float* W_out = (const float*)d_in[3];  // [1024,1024]
  const float* b_out = (const float*)d_in[4];  // [1024]
  const float* wq    = (const float*)d_in[5];  // [16,3,64]
  const float* wvp   = (const float*)d_in[6];  // [16,3,64]
  float* out = (float*)d_out;                  // [8,4096,1024]

  const int Nrows = 8 * 4096;  // 32768
  const int K = 1024, ncols = 1024;

  // Workspace (bf16 elements), total ~208 MB:
  __bf16* wsb   = (__bf16*)d_ws;
  __bf16* qrybf = wsb;               // 33554432 elems: query in bf16
  __bf16* wbf   = qrybf + 33554432;  //  2097152 elems: W_qkv q-rows then v-rows
  __bf16* wobf  = wbf   +  2097152;  //  1048576 elems: W_out
  __bf16* qbuf  = wobf  +  1048576;  // 33554432 elems: q result / wavelet output (in-place)
  __bf16* vbuf  = qbuf  + 33554432;  // 33554432 elems: v result

  dim3 blk(256);
  // one-shot f32 -> bf16 conversions
  cvt_f32_to_bf16_x8<<<dim3(16384), blk, 0, stream>>>(query, qrybf);
  cvt_f32_to_bf16_x8<<<dim3(512),   blk, 0, stream>>>(W_qkv, wbf);                         // rows 0..1023 (q)
  cvt_f32_to_bf16_x8<<<dim3(512),   blk, 0, stream>>>(W_qkv + (size_t)2048 * K, wbf + 1048576); // rows 2048..3071 (v)
  cvt_f32_to_bf16_x8<<<dim3(512),   blk, 0, stream>>>(W_out, wobf);

  dim3 grid(ncols / 128, Nrows / 128);  // (8, 256)
  dwt_gemm_bf16<__bf16><<<grid, blk, 0, stream>>>(qrybf, wbf,           b_qkv,        qbuf, K, ncols);
  dwt_gemm_bf16<__bf16><<<grid, blk, 0, stream>>>(qrybf, wbf + 1048576, b_qkv + 2048, vbuf, K, ncols);
  dwt_haar_mix<<<dim3(16384), blk, 0, stream>>>(qbuf, vbuf, wq, wvp);
  dwt_gemm_bf16<float><<<grid, blk, 0, stream>>>(qbuf, wobf, b_out, out, K, ncols);
}
